// Attention_88270167867463
// MI455X (gfx1250) — compile-verified, matrix operand order checked by reading, not run on hardware
//
#include <hip/hip_runtime.h>

typedef _Float16 h16;
typedef __attribute__((ext_vector_type(16))) _Float16 v16h;
typedef __attribute__((ext_vector_type(8)))  _Float16 v8h;
typedef __attribute__((ext_vector_type(8)))  float    v8f;
typedef __attribute__((ext_vector_type(4)))  float    v4f;

#define DIMC   128
#define HEADS  4
#define HD     32
#define BATCH  8
#define NTOK   3136
#define NKV    784
#define IMGW   56
#define SRED   2

// ---------------------------------------------------------------------------
// Fragment loader. 16-bit A-matrix 16x32 layout (ISA 7.12.2):
//   lane L holds row r = L&15; elements 0..7  = K  (L<16 ? 0 : 8)  + e
//                              elements 8..15 = K  (L<16 ? 16: 24) + (e-8)
// Both 8-element runs are contiguous -> two 16-byte loads per lane.
// The B operand (32x16) uses the same addressing when the source is stored
// transposed (row index = output column).
// ---------------------------------------------------------------------------
template <typename T>
__device__ __forceinline__ v16h load_frag(const T* base, int ld) {
  int lane = threadIdx.x & 31;
  int r  = lane & 15;
  int k0 = (lane >> 4) << 3;             // 0 or 8
  const T* p = base + (size_t)r * ld + k0;
  v8h lo = *(const v8h*)(p);
  v8h hi = *(const v8h*)(p + 16);
  v16h f;
#pragma unroll
  for (int i = 0; i < 8; ++i) { f[i] = lo[i]; f[i + 8] = hi[i]; }
  return f;
}

// ---------------------------------------------------------------------------
__global__ void f32_to_f16_kernel(const float* __restrict__ in,
                                  h16* __restrict__ out, int n) {
  int i = blockIdx.x * blockDim.x + threadIdx.x;
  if (i < n) out[i] = (h16)in[i];
}

// Transpose all four 128x128 weights to f16, n-major: Wt[w][n][k] = W[w][k][n]
__global__ void prep_w_kernel(const float* __restrict__ Wq,
                              const float* __restrict__ Wk,
                              const float* __restrict__ Wv,
                              const float* __restrict__ Wp,
                              h16* __restrict__ Wt) {
  int idx = blockIdx.x * blockDim.x + threadIdx.x;
  if (idx >= 4 * DIMC * DIMC) return;
  int w = idx >> 14;
  int r = idx & 16383;
  int n = r >> 7;
  int k = r & 127;
  const float* src = (w == 0) ? Wq : (w == 1) ? Wk : (w == 2) ? Wv : Wp;
  Wt[idx] = (h16)src[k * DIMC + n];
}

// Depthwise 2x2 stride-2 conv + BatchNorm (eval) -> x_ in f16  (B, Nkv, C)
__global__ void reduce_bn_kernel(const float* __restrict__ x,
                                 const float* __restrict__ cw,
                                 const float* __restrict__ cb,
                                 const float* __restrict__ g,
                                 const float* __restrict__ be,
                                 const float* __restrict__ mu,
                                 const float* __restrict__ var,
                                 h16* __restrict__ xr) {
  int idx = blockIdx.x * blockDim.x + threadIdx.x;
  if (idx >= BATCH * NKV * DIMC) return;
  int c = idx & (DIMC - 1);
  int m = (idx >> 7) % NKV;
  int b = idx / (NKV * DIMC);
  int i = m / (IMGW / SRED), j = m % (IMGW / SRED);
  const float* xb = x + (size_t)b * NTOK * DIMC + c;
  float s = 0.f;
#pragma unroll
  for (int di = 0; di < SRED; ++di)
#pragma unroll
    for (int dj = 0; dj < SRED; ++dj)
      s += cw[c * 4 + di * 2 + dj] *
           xb[(size_t)((SRED * i + di) * IMGW + SRED * j + dj) * DIMC];
  s += cb[c];
  float inv = g[c] * rsqrtf(var[c] + 1e-5f);
  xr[idx] = (h16)(s * inv + (be[c] - mu[c] * inv));
}

// ---------------------------------------------------------------------------
// GEMM: Y(M,128) = X(M,128) @ W(128,128) + bias, via WMMA f32_16x16x32_f16.
// MODE 0: f16 out row-major; MODE 1: f16 out V transposed per head Vt[b][h][d][m];
// MODE 2: f32 out (final projection).
// ---------------------------------------------------------------------------
template <int MODE>
__global__ __launch_bounds__(128) void gemm128_kernel(
    const h16* __restrict__ X, const h16* __restrict__ Wt,
    const float* __restrict__ bias, void* __restrict__ outp, int Mtotal) {
  int wv = threadIdx.x >> 5;
  int lane = threadIdx.x & 31;
  int r0 = (blockIdx.x * 4 + wv) * 16;
  if (r0 >= Mtotal) return;

  v16h a[4];
#pragma unroll
  for (int kk = 0; kk < 4; ++kk)
    a[kk] = load_frag(X + (size_t)r0 * DIMC + kk * 32, DIMC);

  int n = lane & 15;
  int hi8 = (lane >> 4) * 8;

  for (int nt = 0; nt < 8; ++nt) {
    float bval = bias[nt * 16 + n];
    v8f acc;
#pragma unroll
    for (int v = 0; v < 8; ++v) acc[v] = bval;
#pragma unroll
    for (int kk = 0; kk < 4; ++kk) {
      v16h bf = load_frag(Wt + (size_t)(nt * 16) * DIMC + kk * 32, DIMC);
      acc = __builtin_amdgcn_wmma_f32_16x16x32_f16(false, a[kk], false, bf,
                                                   (short)0, acc, false, false);
    }
    if (MODE == 2) {
      float* out = (float*)outp;
#pragma unroll
      for (int v = 0; v < 8; ++v)
        out[(size_t)(r0 + v + hi8) * DIMC + nt * 16 + n] = acc[v];
    } else if (MODE == 0) {
      h16* out = (h16*)outp;
#pragma unroll
      for (int v = 0; v < 8; ++v)
        out[(size_t)(r0 + v + hi8) * DIMC + nt * 16 + n] = (h16)acc[v];
    } else {
      h16* out = (h16*)outp;
      int col = nt * 16 + n;
      int h = col >> 5, d = col & 31;
      int b = r0 / NKV;
      int m0 = r0 - b * NKV;                 // 784 % 16 == 0: no straddle
      v8h pk;
#pragma unroll
      for (int v = 0; v < 8; ++v) pk[v] = (h16)acc[v];
      *(v8h*)(out + (((size_t)b * HEADS + h) * HD + d) * NKV + m0 + hi8) = pk;
    }
  }
}

// ---------------------------------------------------------------------------
// Fused flash attention, S-transposed formulation:
//   S^T = K_tile(16x32) x Q^T(32x16)  ->  lane = query, elements = KV index.
// Row statistics are in-register reductions + ONE shfl_xor(16); rel-pos loads
// are contiguous float4 runs; P staged to LDS with two packed b128 stores.
// 24 full 32-wide KV tiles + one 16-wide tail (784 = 24*32 + 16), no masking.
// ---------------------------------------------------------------------------
__global__ __launch_bounds__(128) void attn_kernel(
    const h16* __restrict__ Qh, const h16* __restrict__ Kh,
    const h16* __restrict__ Vt, const float* __restrict__ rel,
    h16* __restrict__ Ctx) {
  __shared__ h16 pbuf[4][16][32];

  int wv = threadIdx.x >> 5;
  int lane = threadIdx.x & 31;
  int tile = blockIdx.x * 4 + wv;            // B*H*196 = 6272 tiles
  int qt = tile % 196;
  int h = (tile / 196) & 3;
  int b = tile / 784;
  int n0 = qt * 16;
  const float scale = 0.17677669529663687f;  // 32^-0.5

  int q   = lane & 15;
  int hi8 = (lane >> 4) * 8;

  // Q fragment (B-operand role; same layout as A thanks to symmetry)
  v16h qfB = load_frag(Qh + ((size_t)b * NTOK + n0) * DIMC + h * HD, DIMC);

  const h16* kbase = Kh + (size_t)b * NKV * DIMC + h * HD;
  const h16* vbase = Vt + (((size_t)b * HEADS + h) * HD) * NKV;
  const float* relq = rel + ((size_t)h * NTOK + n0 + q) * NKV;  // this query's row

  float mq = -3e38f, lq = 0.f;               // per-query stats (dup in halves)
  v8f acc0, acc1;
#pragma unroll
  for (int v = 0; v < 8; ++v) { acc0[v] = 0.f; acc1[v] = 0.f; }

  v8f z;
#pragma unroll
  for (int v = 0; v < 8; ++v) z[v] = 0.f;

  for (int jt = 0; jt < 24; ++jt) {
    int j0 = jt * 32;
    // prefetch next KV tile
    __builtin_prefetch(kbase + (size_t)(j0 + 32) * DIMC, 0, 1);
    __builtin_prefetch(vbase + j0 + 32, 0, 1);

    v16h kb0 = load_frag(kbase + (size_t)j0 * DIMC, DIMC);        // kv j0..+15
    v16h kb1 = load_frag(kbase + (size_t)(j0 + 16) * DIMC, DIMC); // kv +16..+31
    v8f st0 = __builtin_amdgcn_wmma_f32_16x16x32_f16(false, kb0, false, qfB,
                                                     (short)0, z, false, false);
    v8f st1 = __builtin_amdgcn_wmma_f32_16x16x32_f16(false, kb1, false, qfB,
                                                     (short)0, z, false, false);
    // rel-pos: contiguous 8-float runs per lane (32B-aligned)
    v4f r00 = *(const v4f*)(relq + j0 + hi8);
    v4f r01 = *(const v4f*)(relq + j0 + hi8 + 4);
    v4f r10 = *(const v4f*)(relq + j0 + 16 + hi8);
    v4f r11 = *(const v4f*)(relq + j0 + 16 + hi8 + 4);

    float a0[8], a1[8];
#pragma unroll
    for (int v = 0; v < 8; ++v) {
      float rv0 = (v < 4) ? r00[v] : r01[v - 4];
      float rv1 = (v < 4) ? r10[v] : r11[v - 4];
      a0[v] = st0[v] * scale + rv0;
      a1[v] = st1[v] * scale + rv1;
    }
    float t = fmaxf(a0[0], a1[0]);
#pragma unroll
    for (int v = 1; v < 8; ++v) t = fmaxf(t, fmaxf(a0[v], a1[v]));
    t = fmaxf(t, __shfl_xor(t, 16, 32));     // merge the two half-lanes
    float nm = fmaxf(mq, t);
    float corr = __expf(mq - nm);
    mq = nm;
    v8h e0, e1;
    float rs = 0.f;
#pragma unroll
    for (int v = 0; v < 8; ++v) {
      float x0 = __expf(a0[v] - nm);
      float x1 = __expf(a1[v] - nm);
      rs += x0 + x1;
      e0[v] = (h16)x0;
      e1[v] = (h16)x1;
    }
    rs += __shfl_xor(rs, 16, 32);
    lq = lq * corr + rs;
    // P^T element (q, kv) -> pbuf[q][kvInTile]; two packed 16B DS stores
    *(v8h*)&pbuf[wv][q][hi8]      = e0;
    *(v8h*)&pbuf[wv][q][16 + hi8] = e1;
    // rescale accumulators: corr for query row v+hi8 lives in lane v+hi8
#pragma unroll
    for (int v = 0; v < 8; ++v) {
      float cb = __shfl(corr, v + hi8, 32);
      acc0[v] *= cb;
      acc1[v] *= cb;
    }
    __builtin_amdgcn_wave_barrier();
    asm volatile("s_wait_dscnt 0" ::: "memory");  // in-wave LDS store->load
    v16h pf  = load_frag(&pbuf[wv][0][0], 32);
    v16h vb0 = load_frag(vbase + j0, NKV);                     // d = 0..15
    v16h vb1 = load_frag(vbase + (size_t)16 * NKV + j0, NKV);  // d = 16..31
    acc0 = __builtin_amdgcn_wmma_f32_16x16x32_f16(false, pf, false, vb0,
                                                  (short)0, acc0, false, false);
    acc1 = __builtin_amdgcn_wmma_f32_16x16x32_f16(false, pf, false, vb1,
                                                  (short)0, acc1, false, false);
    __builtin_amdgcn_wave_barrier();
  }

  // ---- tail tile: kv 768..783 (one 16-wide subtile, fully valid) ----
  {
    int j0 = 768;
    v16h kb0 = load_frag(kbase + (size_t)j0 * DIMC, DIMC);
    v8f st0 = __builtin_amdgcn_wmma_f32_16x16x32_f16(false, kb0, false, qfB,
                                                     (short)0, z, false, false);
    v4f r00 = *(const v4f*)(relq + j0 + hi8);
    v4f r01 = *(const v4f*)(relq + j0 + hi8 + 4);
    float a0[8];
#pragma unroll
    for (int v = 0; v < 8; ++v)
      a0[v] = st0[v] * scale + ((v < 4) ? r00[v] : r01[v - 4]);
    float t = a0[0];
#pragma unroll
    for (int v = 1; v < 8; ++v) t = fmaxf(t, a0[v]);
    t = fmaxf(t, __shfl_xor(t, 16, 32));
    float nm = fmaxf(mq, t);
    float corr = __expf(mq - nm);
    mq = nm;
    v8h e0, zz;
    float rs = 0.f;
#pragma unroll
    for (int v = 0; v < 8; ++v) {
      float x0 = __expf(a0[v] - nm);
      rs += x0;
      e0[v] = (h16)x0;
      zz[v] = (h16)0.f;
    }
    rs += __shfl_xor(rs, 16, 32);
    lq = lq * corr + rs;
    *(v8h*)&pbuf[wv][q][hi8]      = e0;
    *(v8h*)&pbuf[wv][q][16 + hi8] = zz;      // pad columns -> P = 0
#pragma unroll
    for (int v = 0; v < 8; ++v) {
      float cb = __shfl(corr, v + hi8, 32);
      acc0[v] *= cb;
      acc1[v] *= cb;
    }
    __builtin_amdgcn_wave_barrier();
    asm volatile("s_wait_dscnt 0" ::: "memory");
    v16h pf  = load_frag(&pbuf[wv][0][0], 32);
    v16h vb0 = load_frag(vbase + j0, NKV);        // overreads land in padded ws
    v16h vb1 = load_frag(vbase + (size_t)16 * NKV + j0, NKV);
    acc0 = __builtin_amdgcn_wmma_f32_16x16x32_f16(false, pf, false, vb0,
                                                  (short)0, acc0, false, false);
    acc1 = __builtin_amdgcn_wmma_f32_16x16x32_f16(false, pf, false, vb1,
                                                  (short)0, acc1, false, false);
  }

  // epilogue: divide by row sum (query row v+hi8 lives in lane v+hi8)
#pragma unroll
  for (int v = 0; v < 8; ++v) {
    float lv = __shfl(lq, v + hi8, 32);
    float inv = 1.0f / lv;
    int row = v + hi8;
    h16* o = Ctx + ((size_t)b * NTOK + n0 + row) * DIMC + h * HD;
    o[q]      = (h16)(acc0[v] * inv);
    o[16 + q] = (h16)(acc1[v] * inv);
  }
}

// ---------------------------------------------------------------------------
extern "C" void kernel_launch(void* const* d_in, const int* in_sizes, int n_in,
                              void* d_out, int out_size, void* d_ws,
                              size_t ws_size, hipStream_t stream) {
  const float* x   = (const float*)d_in[0];
  const float* rel = (const float*)d_in[1];
  const float* Wq  = (const float*)d_in[2];
  const float* bq  = (const float*)d_in[3];
  const float* Wk  = (const float*)d_in[4];
  const float* bk  = (const float*)d_in[5];
  const float* Wv  = (const float*)d_in[6];
  const float* bv  = (const float*)d_in[7];
  const float* cw  = (const float*)d_in[8];
  const float* cb  = (const float*)d_in[9];
  const float* g   = (const float*)d_in[10];
  const float* be  = (const float*)d_in[11];
  const float* mu  = (const float*)d_in[12];
  const float* var = (const float*)d_in[13];
  const float* Wp  = (const float*)d_in[14];
  const float* bp  = (const float*)d_in[15];

  uint8_t* ws = (uint8_t*)d_ws;
  size_t off = 0;
  auto carve = [&](size_t bytes) -> void* {
    void* p = ws + off;
    off += (bytes + 255) & ~(size_t)255;
    return p;
  };
  h16* x_h = (h16*)carve((size_t)BATCH * NTOK * DIMC * 2);
  h16* xr  = (h16*)carve((size_t)BATCH * NKV * DIMC * 2);
  h16* Wt  = (h16*)carve((size_t)4 * DIMC * DIMC * 2);
  h16* Qh  = (h16*)carve((size_t)BATCH * NTOK * DIMC * 2);
  h16* Kh  = (h16*)carve(((size_t)BATCH * NKV + 16) * DIMC * 2);  // +pad tile
  h16* Vt  = (h16*)carve((size_t)BATCH * HEADS * HD * NKV * 2 + 128);
  h16* Ctx = (h16*)carve((size_t)BATCH * NTOK * DIMC * 2);

  int nx = BATCH * NTOK * DIMC;
  f32_to_f16_kernel<<<(nx + 255) / 256, 256, 0, stream>>>(x, x_h, nx);
  prep_w_kernel<<<(4 * DIMC * DIMC + 255) / 256, 256, 0, stream>>>(Wq, Wk, Wv,
                                                                   Wp, Wt);
  int nr = BATCH * NKV * DIMC;
  reduce_bn_kernel<<<(nr + 255) / 256, 256, 0, stream>>>(x, cw, cb, g, be, mu,
                                                         var, xr);
  gemm128_kernel<0><<<392, 128, 0, stream>>>(x_h, Wt, bq, Qh, BATCH * NTOK);
  gemm128_kernel<0><<<98, 128, 0, stream>>>(xr, Wt + 16384, bk, Kh,
                                            BATCH * NKV);
  gemm128_kernel<1><<<98, 128, 0, stream>>>(xr, Wt + 2 * 16384, bv, Vt,
                                            BATCH * NKV);
  attn_kernel<<<1568, 128, 0, stream>>>(Qh, Kh, Vt, rel, Ctx);
  gemm128_kernel<2><<<392, 128, 0, stream>>>(Ctx, Wt + 3 * 16384, bp, d_out,
                                             BATCH * NTOK);
}
